// SwinMSA_57836029608159
// MI455X (gfx1250) — matrix operand order repaired and checked by reading, not compiled, not run on hardware
//
#include <hip/hip_runtime.h>
#include <hip/hip_bf16.h>

// ---------------- problem constants ----------------
#define EMBED     512
#define NHEAD     16
#define NTOK      64
#define HDIM      32
#define NWIN      256
#define BWIN      1024
#define QKV_COLS  1536
#define SCALE_F   0.17677669529663687f   // 32^-0.5

// LDS strides (bf16 elements), padded for bank-conflict-free row access
#define XS 520   // x / attn-out rows (512 + 8)
#define QS 40    // q/k rows (32 + 8)
#define VS 72    // v^T rows (64 + 8)
#define PS 72    // softmaxed P rows (64 + 8)

#define SMEM_X_BYTES   (64 * XS * 2)                  // 66560
#define SMEM_O_BYTES   (64 * XS * 2)                  // 66560
#define SMEM_WAVE_B    (64*QS*2 + 64*QS*2 + 32*VS*2 + 16*PS*2) // 17152
#define SMEM_BYTES     (SMEM_X_BYTES + SMEM_O_BYTES + 8 * SMEM_WAVE_B) // 270336

// workspace layout (bytes)
#define WS_QKVW_OFF 0u
#define WS_QKVW_BYT (QKV_COLS * EMBED * 2u)   // 1,572,864 bf16 [1536][512]
#define WS_PROJ_OFF (WS_QKVW_OFF + WS_QKVW_BYT)
#define WS_PROJ_BYT (EMBED * EMBED * 2u)      //   524,288 bf16 [512][512]
#define WS_BIAS_OFF (WS_PROJ_OFF + WS_PROJ_BYT)
#define WS_BIAS_BYT (NHEAD * NTOK * NTOK * 4u) // 262,144 f32 [16][64][64]

typedef __attribute__((ext_vector_type(16))) __bf16 v16bf;
typedef __attribute__((ext_vector_type(8)))  float  v8f;

__device__ __forceinline__ __bf16 f2bf(float f) {
  union { float f; unsigned u; } x; x.f = f;
  unsigned r = (x.u + 0x7FFFu + ((x.u >> 16) & 1u)) >> 16;
  union { unsigned short s; __bf16 b; } y; y.s = (unsigned short)r;
  return y.b;
}

// A-tile (16x32 bf16, M x K): lane holds row m = lane&15; half-wave splits K.
// ISA layout: lanes0-15 -> K[0..7]=V0-3, K[16..23]=V4-7 ; lanes16-31 -> K[8..15],K[24..31]
__device__ __forceinline__ v16bf load_tileA(const __bf16* base, int stride, int k0) {
  const int lane = threadIdx.x & 31;
  const int m = lane & 15, half = lane >> 4;
  const __bf16* row = base + m * stride + k0 + half * 8;
  union { uint4 u[2]; v16bf v; } r;
  r.u[0] = *reinterpret_cast<const uint4*>(row);
  r.u[1] = *reinterpret_cast<const uint4*>(row + 16);
  return r.v;
}

// B-tile (32x16 bf16, K x N) sourced from a [N][K] (row-per-column) array:
// lane holds column n = lane&15; lanes0-15 -> K[0..7],K[16..23], lanes16-31 -> K[8..15],K[24..31]
__device__ __forceinline__ v16bf load_tileB(const __bf16* base, int stride, int k0) {
  const int lane = threadIdx.x & 31;
  const int n = lane & 15, half = lane >> 4;
  const __bf16* row = base + n * stride + k0 + half * 8;
  union { uint4 u[2]; v16bf v; } r;
  r.u[0] = *reinterpret_cast<const uint4*>(row);
  r.u[1] = *reinterpret_cast<const uint4*>(row + 16);
  return r.v;
}

__device__ __forceinline__ v8f wmma_bf16(v16bf a, v16bf b, v8f c) {
  return __builtin_amdgcn_wmma_f32_16x16x32_bf16(false, a, false, b, (short)0, c, false, false);
}

// ---------------- prep kernels ----------------
// dst[c][k] = bf16(src[k][c])  (src is [K][C] row-major fp32)
__global__ void transpose_conv_kernel(const float* __restrict__ src,
                                      __bf16* __restrict__ dst,
                                      int krows, int cols) {
  int i = blockIdx.x * blockDim.x + threadIdx.x;
  int total = krows * cols;
  if (i >= total) return;
  int c = i / krows;
  int k = i - c * krows;
  dst[i] = f2bf(src[(size_t)k * cols + c]);
}

// biasM[h][i][j] = bias_table[rel_index[i*64+j]][h]
__global__ void bias_gather_kernel(const float* __restrict__ table,
                                   const int* __restrict__ relidx,
                                   float* __restrict__ dst) {
  int i = blockIdx.x * blockDim.x + threadIdx.x;
  if (i >= NHEAD * NTOK * NTOK) return;
  int h = i >> 12;
  int ij = i & 4095;
  dst[i] = table[relidx[ij] * NHEAD + h];
}

// ---------------- fused window-attention kernel ----------------
__global__ __launch_bounds__(256) void swin_fused_kernel(
    const float* __restrict__ x,          // [1024][64][512]
    const float* __restrict__ mask,       // [256][64][64]
    const float* __restrict__ qkv_b,      // [1536]
    const float* __restrict__ proj_b,     // [512]
    const __bf16* __restrict__ qkvWt,     // [1536][512]
    const __bf16* __restrict__ projWt,    // [512][512]
    const float* __restrict__ biasM,      // [16][64][64]
    float* __restrict__ out)              // [1024][64][512]
{
  extern __shared__ __align__(16) char smem[];
  __bf16* xs   = reinterpret_cast<__bf16*>(smem);
  __bf16* outb = reinterpret_cast<__bf16*>(smem + SMEM_X_BYTES);
  const int tid  = threadIdx.x;
  const int lane = tid & 31;
  const int wave = tid >> 5;
  const int l15  = lane & 15;
  const int half = lane >> 4;
  const int blk  = blockIdx.x;
  const int win  = blk & (NWIN - 1);

  char* wbase = smem + SMEM_X_BYTES + SMEM_O_BYTES + wave * SMEM_WAVE_B;
  __bf16* qbuf = reinterpret_cast<__bf16*>(wbase);
  __bf16* kbuf = reinterpret_cast<__bf16*>(wbase + 64 * QS * 2);
  __bf16* vtb  = reinterpret_cast<__bf16*>(wbase + 2 * 64 * QS * 2);
  __bf16* pbuf = reinterpret_cast<__bf16*>(wbase + 2 * 64 * QS * 2 + 32 * VS * 2);

  // ---- phase 0: stage x tile (64x512) into LDS as bf16 ----
  // float4 global loads (b128, coalesced) -> packed 4xbf16 ds_store_b64
  {
    const float4* xg4 =
        reinterpret_cast<const float4*>(x + (size_t)blk * NTOK * EMBED);
    for (int i = tid; i < NTOK * EMBED / 4; i += 256) {
      const float4 f = xg4[i];
      const int t = i >> 7;            // 128 float4 per 512-wide row
      const int d = (i & 127) * 4;
      union { __bf16 b[4]; uint2 u; } p;
      p.b[0] = f2bf(f.x); p.b[1] = f2bf(f.y);
      p.b[2] = f2bf(f.z); p.b[3] = f2bf(f.w);
      *reinterpret_cast<uint2*>(&xs[t * XS + d]) = p.u;
    }
  }
  __syncthreads();

  // ---- phases 1-3: per-wave, 2 heads each: QKV -> attention -> PV ----
  for (int hh = 0; hh < 2; ++hh) {
    const int h = wave + hh * 8;

    // QKV GEMM for this head: q,k,v are 64x32, K = 512
    #pragma unroll
    for (int mat = 0; mat < 3; ++mat) {
      const int colbase = h * 96 + mat * 32;
      #pragma unroll
      for (int nt = 0; nt < 2; ++nt) {
        const int n0 = colbase + nt * 16;
        v8f acc0 = {}, acc1 = {}, acc2 = {}, acc3 = {};
        const __bf16* wrow = qkvWt + (size_t)n0 * EMBED;
        for (int ks = 0; ks < 16; ++ks) {
          v16bf B = load_tileB(wrow, EMBED, ks * 32);
          acc0 = wmma_bf16(load_tileA(xs +  0 * XS, XS, ks * 32), B, acc0);
          acc1 = wmma_bf16(load_tileA(xs + 16 * XS, XS, ks * 32), B, acc1);
          acc2 = wmma_bf16(load_tileA(xs + 32 * XS, XS, ks * 32), B, acc2);
          acc3 = wmma_bf16(load_tileA(xs + 48 * XS, XS, ks * 32), B, acc3);
        }
        const float bb = qkv_b[n0 + l15];
        const int d = nt * 16 + l15;
        v8f accs[4] = {acc0, acc1, acc2, acc3};
        #pragma unroll
        for (int mt = 0; mt < 4; ++mt) {
          #pragma unroll
          for (int r = 0; r < 8; ++r) {
            const float v = accs[mt][r] + bb;
            const int row = mt * 16 + r + 8 * half;
            if (mat == 0)      qbuf[row * QS + d] = f2bf(v);
            else if (mat == 1) kbuf[row * QS + d] = f2bf(v);
            else               vtb[d * VS + row]  = f2bf(v);   // store V transposed
          }
        }
      }
    }

    // attention for this head, one 16-row strip at a time
    const float* biasH = biasM + (size_t)h * NTOK * NTOK;
    const float* maskW = mask + (size_t)win * NTOK * NTOK;
    #pragma unroll 1
    for (int mt = 0; mt < 4; ++mt) {
      v16bf Aq = load_tileA(qbuf + mt * 16 * QS, QS, 0);
      v8f s[4];
      #pragma unroll
      for (int t = 0; t < 4; ++t) {
        v8f z = {};
        s[t] = wmma_bf16(Aq, load_tileB(kbuf + t * 16 * QS, QS, 0), z);
      }
      // scale + relative-position bias + window mask
      #pragma unroll
      for (int t = 0; t < 4; ++t) {
        const int col = t * 16 + l15;
        #pragma unroll
        for (int r = 0; r < 8; ++r) {
          const int row = mt * 16 + r + 8 * half;
          s[t][r] = s[t][r] * SCALE_F + biasH[row * 64 + col] + maskW[row * 64 + col];
        }
      }
      // row softmax: each row's 64 values live in one 16-lane half
      #pragma unroll
      for (int r = 0; r < 8; ++r) {
        float mx = fmaxf(fmaxf(s[0][r], s[1][r]), fmaxf(s[2][r], s[3][r]));
        #pragma unroll
        for (int off = 1; off < 16; off <<= 1) mx = fmaxf(mx, __shfl_xor(mx, off, 32));
        float e0 = __expf(s[0][r] - mx), e1 = __expf(s[1][r] - mx);
        float e2 = __expf(s[2][r] - mx), e3 = __expf(s[3][r] - mx);
        float su = e0 + e1 + e2 + e3;
        #pragma unroll
        for (int off = 1; off < 16; off <<= 1) su += __shfl_xor(su, off, 32);
        const float inv = 1.0f / su;
        const int prow = (r + 8 * half) * PS + l15;
        pbuf[prow +  0] = f2bf(e0 * inv);
        pbuf[prow + 16] = f2bf(e1 * inv);
        pbuf[prow + 32] = f2bf(e2 * inv);
        pbuf[prow + 48] = f2bf(e3 * inv);
      }
      // PV: (16x64) @ (64x32) -> strip of attn output
      #pragma unroll
      for (int dt = 0; dt < 2; ++dt) {
        v8f acc = {};
        #pragma unroll
        for (int kt = 0; kt < 2; ++kt) {
          acc = wmma_bf16(load_tileA(pbuf, PS, kt * 32),
                          load_tileB(vtb + dt * 16 * VS, VS, kt * 32), acc);
        }
        const int col = h * 32 + dt * 16 + l15;
        #pragma unroll
        for (int r = 0; r < 8; ++r)
          outb[(mt * 16 + r + 8 * half) * XS + col] = f2bf(acc[r]);
      }
    }
  }
  __syncthreads();

  // ---- phase 4: output projection (64x512) @ (512x512) + bias ----
  for (int idx = wave; idx < 128; idx += 8) {
    const int mt = idx >> 5, nt = idx & 31;
    v8f acc = {};
    const __bf16* wrow = projWt + (size_t)(nt * 16) * EMBED;
    for (int ks = 0; ks < 16; ++ks) {
      acc = wmma_bf16(load_tileA(outb + mt * 16 * XS, XS, ks * 32),
                      load_tileB(wrow, EMBED, ks * 32), acc);
    }
    const float pb = proj_b[nt * 16 + l15];
    const int col = nt * 16 + l15;
    #pragma unroll
    for (int r = 0; r < 8; ++r) {
      const int tok = blk * NTOK + mt * 16 + r + 8 * half;
      out[(size_t)tok * EMBED + col] = acc[r] + pb;
    }
  }
}

// ---------------- host launcher ----------------
extern "C" void kernel_launch(void* const* d_in, const int* in_sizes, int n_in,
                              void* d_out, int out_size, void* d_ws, size_t ws_size,
                              hipStream_t stream) {
  (void)in_sizes; (void)n_in; (void)out_size; (void)ws_size;
  const float* x        = (const float*)d_in[0];
  const float* attn_msk = (const float*)d_in[1];
  const float* qkv_w    = (const float*)d_in[2];
  const float* qkv_b    = (const float*)d_in[3];
  const float* proj_w   = (const float*)d_in[4];
  const float* proj_b   = (const float*)d_in[5];
  const float* btable   = (const float*)d_in[6];
  const int*   relidx   = (const int*)d_in[7];
  float* out = (float*)d_out;

  char* ws = (char*)d_ws;
  __bf16* qkvWt  = reinterpret_cast<__bf16*>(ws + WS_QKVW_OFF);
  __bf16* projWt = reinterpret_cast<__bf16*>(ws + WS_PROJ_OFF);
  float*  biasM  = reinterpret_cast<float*>(ws + WS_BIAS_OFF);

  // prep: bf16 transposed weights + gathered bias (stream-ordered before main)
  {
    int total = EMBED * QKV_COLS;
    transpose_conv_kernel<<<(total + 255) / 256, 256, 0, stream>>>(qkv_w, qkvWt, EMBED, QKV_COLS);
  }
  {
    int total = EMBED * EMBED;
    transpose_conv_kernel<<<(total + 255) / 256, 256, 0, stream>>>(proj_w, projWt, EMBED, EMBED);
  }
  {
    int total = NHEAD * NTOK * NTOK;
    bias_gather_kernel<<<(total + 255) / 256, 256, 0, stream>>>(btable, relidx, biasM);
  }

  // deterministic, unconditional (no static guards): not a stream op, safe
  // under graph capture; ignore return (ROCm may allow >64KB without opt-in).
  (void)hipFuncSetAttribute(reinterpret_cast<const void*>(swin_fused_kernel),
                            hipFuncAttributeMaxDynamicSharedMemorySize, SMEM_BYTES);

  swin_fused_kernel<<<BWIN, 256, SMEM_BYTES, stream>>>(
      x, attn_msk, qkv_b, proj_b, qkvWt, projWt, biasM, out);
}
